// DiTBlock_9380208575227
// MI455X (gfx1250) — compile-verified, matrix operand order checked by reading, not run on hardware
//
#include <hip/hip_runtime.h>
#include <hip/hip_bf16.h>

#define S_LEN 1024
#define DMODEL 1024
#define FDIM 4096
#define NHEAD 16
#define DHEAD 64
#define BATCH 4
#define MTOT (BATCH * S_LEN) // 4096

typedef __bf16 bf16;
typedef __attribute__((ext_vector_type(16))) __bf16 v16bf;
typedef __attribute__((ext_vector_type(8)))  __bf16 v8bf;
typedef __attribute__((ext_vector_type(8)))  float  v8f;
typedef __attribute__((ext_vector_type(4)))  unsigned int u32x4;
typedef __attribute__((ext_vector_type(8)))  int i32x8;
typedef __attribute__((ext_vector_type(4)))  int i32x4;

__device__ __forceinline__ v8f zero8() {
    v8f z;
#pragma unroll
    for (int i = 0; i < 8; ++i) z[i] = 0.f;
    return z;
}

// Load one 16-element bf16 fragment half-pair: 8 elems at p, 8 at p+16.
// Matches CDNA5 A/B fragment layout when p = row_base + (lane>=16 ? 8 : 0).
__device__ __forceinline__ v16bf frag_ld(const bf16* p) {
    v8bf a = *(const v8bf*)p;
    v8bf b = *(const v8bf*)(p + 16);
    v16bf r;
#pragma unroll
    for (int i = 0; i < 8; ++i) { r[i] = a[i]; r[8 + i] = b[i]; }
    return r;
}

__device__ __forceinline__ float siluf_(float x) { return x / (1.f + __expf(-x)); }

// ---------------------------------------------------------------------------
// TDM: async 2D bf16 tile load (32 x tile_d1 elems) from global to LDS,
// with LDS padding of 16B after every 64B row chunk (ISA 08_async_tensor §8).
// Padded LDS row stride = 80B = 40 bf16 (bank-conflict-free fragment reads).
// ---------------------------------------------------------------------------
__device__ __forceinline__ void tdm_load_tile32(unsigned lds_off, const void* gaddr,
                                                unsigned tile_d1,
                                                unsigned tens_d0, unsigned tens_d1,
                                                unsigned stride0) {
    unsigned long long ga = (unsigned long long)(size_t)gaddr;
    u32x4 g0;
    g0[0] = 1u;                                   // count=1, user descriptor
    g0[1] = lds_off;                              // LDS byte address
    g0[2] = (unsigned)(ga & 0xFFFFFFFFull);       // global_addr[31:0]
    g0[3] = (unsigned)((ga >> 32) & 0x1FFFFFFull) // global_addr[56:32]
            | (2u << 30);                         // type=2 ("image")
    // dw0: data_size=1 (2B) | pad_enable | pad_interval=3 (8x8B=64B) |
    //      pad_amount=3 (4 dwords = 16B)
    const unsigned cfg = (1u << 16) | (1u << 20) | (3u << 22) | (3u << 25);
    i32x8 g1;
    g1[0] = (int)cfg;
    g1[1] = (int)((tens_d0 & 0xFFFFu) << 16);                     // tensor_dim0 lo
    g1[2] = (int)(((tens_d0 >> 16) & 0xFFFFu) | ((tens_d1 & 0xFFFFu) << 16));
    g1[3] = (int)(((tens_d1 >> 16) & 0xFFFFu) | (32u << 16));     // tile_dim0=32
    g1[4] = (int)(tile_d1 & 0xFFFFu);                             // tile_dim1
    g1[5] = (int)stride0;                                         // dim0 stride lo32
    g1[6] = 0;
    g1[7] = 0;
    i32x4 z4;
    z4[0] = 0; z4[1] = 0; z4[2] = 0; z4[3] = 0;
    i32x8 z8;
#pragma unroll
    for (int i = 0; i < 8; ++i) z8[i] = 0;
    __builtin_amdgcn_tensor_load_to_lds(g0, g1, z4, z4, z8, 0);
}

// ---------------------------------------------------------------------------
// Weight transpose+convert: w[K][N] f32  ->  wt[N][K] bf16
// ---------------------------------------------------------------------------
__global__ void transpose_w_kernel(const float* __restrict__ w, bf16* __restrict__ wt,
                                   int K, int N) {
    __shared__ float t[32][33];
    int bx = blockIdx.x * 32, by = blockIdx.y * 32; // bx over N, by over K
#pragma unroll
    for (int i = 0; i < 4; ++i)
        t[threadIdx.y + i * 8][threadIdx.x] =
            w[(size_t)(by + threadIdx.y + i * 8) * N + bx + threadIdx.x];
    __syncthreads();
#pragma unroll
    for (int i = 0; i < 4; ++i)
        wt[(size_t)(bx + threadIdx.y + i * 8) * K + by + threadIdx.x] =
            (bf16)t[threadIdx.x][threadIdx.y + i * 8];
}

// ---------------------------------------------------------------------------
// V[b*S+s][h*64+d] bf16 -> Vt[((b*H+h)*64+d)][s] bf16
// ---------------------------------------------------------------------------
__global__ void vtranspose_kernel(const bf16* __restrict__ V, bf16* __restrict__ Vt) {
    __shared__ bf16 t[32][33];
    int bh = blockIdx.z;
    int b = bh >> 4, h = bh & 15;
    int s0 = blockIdx.x * 32, d0 = blockIdx.y * 32;
#pragma unroll
    for (int i = 0; i < 4; ++i)
        t[threadIdx.y + i * 8][threadIdx.x] =
            V[(size_t)(b * S_LEN + s0 + threadIdx.y + i * 8) * DMODEL + h * DHEAD + d0 + threadIdx.x];
    __syncthreads();
#pragma unroll
    for (int i = 0; i < 4; ++i)
        Vt[(size_t)(bh * DHEAD + d0 + threadIdx.y + i * 8) * S_LEN + s0 + threadIdx.x] =
            t[threadIdx.x][threadIdx.y + i * 8];
}

// ---------------------------------------------------------------------------
// ada[b][c] = ada_b[c] + sum_k silu(t_emb[b][k]) * ada_w[k][c]   (tiny GEMM)
// ---------------------------------------------------------------------------
__global__ void ada_kernel(const float* __restrict__ t_emb, const float* __restrict__ ada_w,
                           const float* __restrict__ ada_b, float* __restrict__ ada) {
    int idx = blockIdx.x * blockDim.x + threadIdx.x; // BATCH * 6*DMODEL
    int b = idx / (6 * DMODEL), c = idx % (6 * DMODEL);
    float acc = ada_b[c];
    for (int k = 0; k < DMODEL; ++k)
        acc += siluf_(t_emb[b * DMODEL + k]) * ada_w[(size_t)k * (6 * DMODEL) + c];
    ada[idx] = acc;
}

// ---------------------------------------------------------------------------
// out_bf16[row][c] = ( LN(x[row]) * g + b ) * (1 + ada_sc[b][c]) + ada_sh[b][c]
// ---------------------------------------------------------------------------
__global__ void ln_mod_kernel(const float* __restrict__ x, const float* __restrict__ gln,
                              const float* __restrict__ bln, const float* __restrict__ ada,
                              int sh_ofs, int sc_ofs, bf16* __restrict__ out) {
    int row = blockIdx.x, tid = threadIdx.x;
    int b = row >> 10;
    const float* xr = x + (size_t)row * DMODEL;
    __shared__ float s1[256], s2[256];
    float a0 = 0.f, a1 = 0.f;
#pragma unroll
    for (int i = 0; i < 4; ++i) { float v = xr[tid + i * 256]; a0 += v; a1 += v * v; }
    s1[tid] = a0; s2[tid] = a1;
    __syncthreads();
    for (int s = 128; s > 0; s >>= 1) {
        if (tid < s) { s1[tid] += s1[tid + s]; s2[tid] += s2[tid + s]; }
        __syncthreads();
    }
    float mu = s1[0] * (1.f / DMODEL);
    float var = s2[0] * (1.f / DMODEL) - mu * mu;
    float rstd = rsqrtf(var + 1e-6f);
#pragma unroll
    for (int i = 0; i < 4; ++i) {
        int c = tid + i * 256;
        float v = (xr[c] - mu) * rstd * gln[c] + bln[c];
        float sc = ada[b * (6 * DMODEL) + sc_ofs + c];
        float sh = ada[b * (6 * DMODEL) + sh_ofs + c];
        out[(size_t)row * DMODEL + c] = (bf16)(v * (1.f + sc) + sh);
    }
}

// ---------------------------------------------------------------------------
// WMMA GEMM: C[M][N] = A[M][K] (bf16, row-major) x Bt[N][K] (bf16, row-major)
// MODE 0: f32 out          MODE 1: bf16 out
// MODE 2: bf16 out = silu(aux) * acc      (SwiGLU fuse)
// MODE 3: f32 out = res + ada_g[b][col] * acc   (AdaLN-gated residual)
// Block tile 128x128, 8 waves of 64x32. TDM (tensor_load_to_lds) streams the
// 128x32 A/B tiles into double-buffered padded LDS, wave 0 drives the DMA and
// synchronizes with s_wait_tensorcnt; compute overlaps the next stage's DMA.
// ---------------------------------------------------------------------------
template <int MODE>
__global__ __launch_bounds__(256) void gemm_kernel(
    const bf16* __restrict__ A, const bf16* __restrict__ Bt, void* __restrict__ Cout,
    const float* __restrict__ aux, const float* __restrict__ res,
    const float* __restrict__ ada, int gofs, int M, int N, int K) {
    constexpr int KT = 32;
    constexpr int ST = 40; // padded LDS row stride (80B) -> bank-conflict-free
    __shared__ bf16 As[2][128 * ST];
    __shared__ bf16 Bs[2][128 * ST];
    int tid = threadIdx.x;
    int lane = tid & 31, w = tid >> 5;
    int wr = w >> 2, wc = w & 3;      // 2x4 wave grid
    int m0 = blockIdx.y * 128, n0 = blockIdx.x * 128;
    int hi8 = (lane >> 4) * 8;
    int mrow = lane & 15;

    v8f c[4][2];
#pragma unroll
    for (int i = 0; i < 4; ++i)
#pragma unroll
        for (int j = 0; j < 2; ++j) c[i][j] = zero8();

    int nk = K / KT;
    if (w == 0) { // preload stage 0
        tdm_load_tile32((unsigned)(size_t)&As[0][0], A + (size_t)m0 * K, 128, K, M, K);
        tdm_load_tile32((unsigned)(size_t)&Bs[0][0], Bt + (size_t)n0 * K, 128, K, N, K);
    }
    for (int i = 0; i < nk; ++i) {
        int buf = i & 1;
        if (w == 0) {
            if (i + 1 < nk) { // issue next stage, then wait for current pair
                tdm_load_tile32((unsigned)(size_t)&As[buf ^ 1][0],
                                A + (size_t)m0 * K + (i + 1) * KT, 128, K, M, K);
                tdm_load_tile32((unsigned)(size_t)&Bs[buf ^ 1][0],
                                Bt + (size_t)n0 * K + (i + 1) * KT, 128, K, N, K);
                __builtin_amdgcn_s_wait_tensorcnt(2);
            } else {
                __builtin_amdgcn_s_wait_tensorcnt(0);
            }
        }
        __syncthreads(); // stage i tiles visible to all waves

        v16bf af[4], bq[2];
#pragma unroll
        for (int mf = 0; mf < 4; ++mf)
            af[mf] = frag_ld(&As[buf][(wr * 64 + mf * 16 + mrow) * ST + hi8]);
#pragma unroll
        for (int nf = 0; nf < 2; ++nf)
            bq[nf] = frag_ld(&Bs[buf][(wc * 32 + nf * 16 + mrow) * ST + hi8]);
#pragma unroll
        for (int mf = 0; mf < 4; ++mf)
#pragma unroll
            for (int nf = 0; nf < 2; ++nf)
                c[mf][nf] = __builtin_amdgcn_wmma_f32_16x16x32_bf16(
                    false, af[mf], false, bq[nf], (short)0, c[mf][nf], false, false);
        __syncthreads(); // buffer reusable for stage i+2
    }

    int nloc = lane & 15, hif = lane >> 4;
#pragma unroll
    for (int mf = 0; mf < 4; ++mf)
#pragma unroll
        for (int nf = 0; nf < 2; ++nf)
#pragma unroll
            for (int r = 0; r < 8; ++r) {
                int row = m0 + wr * 64 + mf * 16 + hif * 8 + r;
                int col = n0 + wc * 32 + nf * 16 + nloc;
                size_t idx = (size_t)row * N + col;
                float v = c[mf][nf][r];
                if (MODE == 0) {
                    ((float*)Cout)[idx] = v;
                } else if (MODE == 1) {
                    ((bf16*)Cout)[idx] = (bf16)v;
                } else if (MODE == 2) {
                    float a0 = aux[idx];
                    ((bf16*)Cout)[idx] = (bf16)(siluf_(a0) * v);
                } else {
                    float gg = ada[(row >> 10) * (6 * DMODEL) + gofs + col];
                    ((float*)Cout)[idx] = res[idx] + gg * v;
                }
            }
}

// ---------------------------------------------------------------------------
// Flash attention, one wave per (b, h, 16-query tile); branch-specific bias
// computed analytically. Q:[b*S+s][h*64+d], K same, Vt:[(b*H+h)*64+d][s].
// ---------------------------------------------------------------------------
__global__ __launch_bounds__(256) void flash_kernel(
    const bf16* __restrict__ Q, const bf16* __restrict__ Kmat,
    const bf16* __restrict__ Vt, bf16* __restrict__ O, int branch) {
    constexpr int PST = 40;
    __shared__ bf16 P[8][16 * PST]; // per-wave P-tile scratch (C->A layout swap)
    int tid = threadIdx.x, lane = tid & 31, w = tid >> 5;
    int task = blockIdx.x * 8 + w;            // 4096 tasks
    int qt = task & 63, h = (task >> 6) & 15, b = task >> 10;
    int q0 = qt * 16;
    int hif = lane >> 4, nloc = lane & 15, hi8 = hif * 8;

    const bf16* qb = Q + (size_t)(b * S_LEN + q0 + nloc) * DMODEL + h * DHEAD;
    v16bf aq[2];
#pragma unroll
    for (int kf = 0; kf < 2; ++kf) aq[kf] = frag_ld(qb + kf * 32 + hi8);

    v8f o[4];
#pragma unroll
    for (int vf = 0; vf < 4; ++vf) o[vf] = zero8();
    float rm[8], rs[8];
#pragma unroll
    for (int r = 0; r < 8; ++r) { rm[r] = -1e30f; rs[r] = 0.f; }

    int kb_lo = 0, kb_hi = (S_LEN / 32) - 1;
    if (branch == 1) { // band |q-k|<=1
        kb_lo = (q0 > 0 ? (q0 - 1) : 0) >> 5;
        int kmax = q0 + 16; if (kmax > S_LEN - 1) kmax = S_LEN - 1;
        kb_hi = kmax >> 5;
    } else if (branch == 2) { // 2x2 entangled pairs stay in the q-tile's block
        kb_lo = kb_hi = q0 >> 5;
    }

    for (int kb = kb_lo; kb <= kb_hi; ++kb) {
        int kbase = kb * 32;
        v8f s[2];
        s[0] = zero8(); s[1] = zero8();
#pragma unroll
        for (int nf = 0; nf < 2; ++nf) {
            const bf16* kp = Kmat + (size_t)(b * S_LEN + kbase + nf * 16 + nloc) * DMODEL + h * DHEAD;
#pragma unroll
            for (int kf = 0; kf < 2; ++kf) {
                v16bf bk = frag_ld(kp + kf * 32 + hi8);
                s[nf] = __builtin_amdgcn_wmma_f32_16x16x32_bf16(
                    false, aq[kf], false, bk, (short)0, s[nf], false, false);
            }
        }
        float corr[8];
#pragma unroll
        for (int r = 0; r < 8; ++r) {
            int q = q0 + hif * 8 + r;
            float mx = -1e30f;
#pragma unroll
            for (int nf = 0; nf < 2; ++nf) {
                int k = kbase + nf * 16 + nloc;
                int dd = q - k; if (dd < 0) dd = -dd;
                float bias;
                if (branch == 0) bias = -(float)dd;
                else if (branch == 1) bias = (dd <= 1) ? 0.f : -1e9f;
                else if (branch == 2) bias = ((q >> 1) == (k >> 1)) ? 0.f : -1e9f;
                else bias = 0.f;
                float sv = s[nf][r] * 0.125f + bias; // 1/sqrt(64)
                s[nf][r] = sv;
                mx = fmaxf(mx, sv);
            }
#pragma unroll
            for (int m = 1; m < 16; m <<= 1) mx = fmaxf(mx, __shfl_xor(mx, m, 16));
            float nm = fmaxf(rm[r], mx);
            float cr = __expf(rm[r] - nm);
            rm[r] = nm; corr[r] = cr;
            float psum = 0.f;
#pragma unroll
            for (int nf = 0; nf < 2; ++nf) {
                float p = __expf(s[nf][r] - nm);
                s[nf][r] = p; psum += p;
            }
#pragma unroll
            for (int m = 1; m < 16; m <<= 1) psum += __shfl_xor(psum, m, 16);
            rs[r] = rs[r] * cr + psum;
        }
#pragma unroll
        for (int vf = 0; vf < 4; ++vf)
#pragma unroll
            for (int r = 0; r < 8; ++r) o[vf][r] *= corr[r];
        // C-layout -> A-layout swap through per-wave LDS tile
#pragma unroll
        for (int nf = 0; nf < 2; ++nf)
#pragma unroll
            for (int r = 0; r < 8; ++r)
                P[w][(hif * 8 + r) * PST + nf * 16 + nloc] = (bf16)s[nf][r];
        asm volatile("s_wait_dscnt 0" ::: "memory"); // LDS stores visible in-wave
        v16bf pa = frag_ld(&P[w][nloc * PST + hi8]);
#pragma unroll
        for (int vf = 0; vf < 4; ++vf) {
            const bf16* vp = Vt + (size_t)((b * NHEAD + h) * DHEAD + vf * 16 + nloc) * S_LEN + kbase;
            v16bf bv = frag_ld(vp + hi8);
            o[vf] = __builtin_amdgcn_wmma_f32_16x16x32_bf16(
                false, pa, false, bv, (short)0, o[vf], false, false);
        }
    }
#pragma unroll
    for (int vf = 0; vf < 4; ++vf)
#pragma unroll
        for (int r = 0; r < 8; ++r)
            O[(size_t)(b * S_LEN + q0 + hif * 8 + r) * DMODEL + h * DHEAD + vf * 16 + nloc] =
                (bf16)(o[vf][r] / rs[r]);
}

// ---------------------------------------------------------------------------
// Per-token gate softmax over 4 branches + fused sum (bf16 out).
// ---------------------------------------------------------------------------
__global__ void gatefuse_kernel(const bf16* __restrict__ o0, const bf16* __restrict__ o1,
                                const bf16* __restrict__ o2, const bf16* __restrict__ o3,
                                const float* __restrict__ wg, const float* __restrict__ bg,
                                bf16* __restrict__ fused) {
    int t = blockIdx.x, tid = threadIdx.x;
    const bf16* ob[4] = { o0 + (size_t)t * DMODEL, o1 + (size_t)t * DMODEL,
                          o2 + (size_t)t * DMODEL, o3 + (size_t)t * DMODEL };
    __shared__ float red[4 * 256];
    __shared__ float gate[4];
    float lacc[4] = {0.f, 0.f, 0.f, 0.f};
    for (int d = tid; d < DMODEL; d += 256) {
#pragma unroll
        for (int j = 0; j < 4; ++j) {
            float ov = (float)ob[j][d];
            const float* wrow = wg + (size_t)(j * DMODEL + d) * 4;
#pragma unroll
            for (int jj = 0; jj < 4; ++jj) lacc[jj] += ov * wrow[jj];
        }
    }
#pragma unroll
    for (int jj = 0; jj < 4; ++jj) red[jj * 256 + tid] = lacc[jj];
    __syncthreads();
    for (int s = 128; s > 0; s >>= 1) {
        if (tid < s)
#pragma unroll
            for (int jj = 0; jj < 4; ++jj) red[jj * 256 + tid] += red[jj * 256 + tid + s];
        __syncthreads();
    }
    if (tid == 0) {
        float l[4], mx = -1e30f;
#pragma unroll
        for (int jj = 0; jj < 4; ++jj) { l[jj] = red[jj * 256] + bg[jj]; mx = fmaxf(mx, l[jj]); }
        float sum = 0.f;
#pragma unroll
        for (int jj = 0; jj < 4; ++jj) { l[jj] = __expf(l[jj] - mx); sum += l[jj]; }
#pragma unroll
        for (int jj = 0; jj < 4; ++jj) gate[jj] = l[jj] / sum;
    }
    __syncthreads();
    for (int d = tid; d < DMODEL; d += 256) {
        float f = gate[0] * (float)ob[0][d] + gate[1] * (float)ob[1][d] +
                  gate[2] * (float)ob[2][d] + gate[3] * (float)ob[3][d];
        fused[(size_t)t * DMODEL + d] = (bf16)f;
    }
}

// ---------------------------------------------------------------------------
extern "C" void kernel_launch(void* const* d_in, const int* in_sizes, int n_in,
                              void* d_out, int out_size, void* d_ws, size_t ws_size,
                              hipStream_t stream) {
    (void)in_sizes; (void)n_in; (void)out_size; (void)ws_size;
    const float* x      = (const float*)d_in[0];
    const float* t_emb  = (const float*)d_in[1];
    const float* ln1_g  = (const float*)d_in[2];
    const float* ln1_b  = (const float*)d_in[3];
    const float* ln2_g  = (const float*)d_in[4];
    const float* ln2_b  = (const float*)d_in[5];
    const float* ada_w  = (const float*)d_in[6];
    const float* ada_bi = (const float*)d_in[7];
    const float* fus_wg = (const float*)d_in[8];
    const float* fus_bg = (const float*)d_in[9];
    const float* fus_wo = (const float*)d_in[10];
    const float* ffn_w1 = (const float*)d_in[11];
    const float* ffn_w2 = (const float*)d_in[12];
    const float* ffn_w3 = (const float*)d_in[13];
    float* out = (float*)d_out;

    char* ws = (char*)d_ws;
    size_t off = 0;
    auto alloc = [&](size_t bytes) -> char* {
        char* p = ws + off;
        off += (bytes + 255) & ~(size_t)255;
        return p;
    };
    const size_t MB1 = (size_t)1024 * 1024; // elems in a 1024x1024 matrix
    bf16*  wT     = (bf16*)alloc(16 * MB1 * 2);      // 16 attn weights, [N][K] bf16
    bf16*  fuswoT = (bf16*)alloc(MB1 * 2);
    bf16*  w1T    = (bf16*)alloc(4 * MB1 * 2);       // [F][D]
    bf16*  w3T    = (bf16*)alloc(4 * MB1 * 2);       // [F][D]
    bf16*  w2T    = (bf16*)alloc(4 * MB1 * 2);       // [D][F]
    float* ada    = (float*)alloc((size_t)BATCH * 6 * DMODEL * 4);
    bf16*  nx     = (bf16*)alloc(4 * MB1 * 2);       // [4096][1024]
    bf16*  Qb     = (bf16*)alloc(4 * MB1 * 2);
    bf16*  Kb     = (bf16*)alloc(4 * MB1 * 2);
    bf16*  Vb     = (bf16*)alloc(4 * MB1 * 2);
    bf16*  Vtb    = (bf16*)alloc(4 * MB1 * 2);
    bf16*  attn   = (bf16*)alloc(4 * MB1 * 2);
    bf16*  obr[4];
    for (int p = 0; p < 4; ++p) obr[p] = (bf16*)alloc(4 * MB1 * 2);
    bf16*  fused  = (bf16*)alloc(4 * MB1 * 2);
    float* x1     = (float*)alloc(4 * MB1 * 4);
    bf16*  nx2    = (bf16*)alloc(4 * MB1 * 2);
    float* h1     = (float*)alloc(16 * MB1 * 4);     // [4096][4096] f32
    bf16*  hh     = (bf16*)alloc(16 * MB1 * 2);

    dim3 tb(32, 8);
    // Weight conversion/transpose to bf16 [N][K]
    for (int j = 0; j < 16; ++j)
        transpose_w_kernel<<<dim3(32, 32), tb, 0, stream>>>(
            (const float*)d_in[14 + j], wT + (size_t)j * MB1, DMODEL, DMODEL);
    transpose_w_kernel<<<dim3(32, 32), tb, 0, stream>>>(fus_wo, fuswoT, DMODEL, DMODEL);
    transpose_w_kernel<<<dim3(128, 32), tb, 0, stream>>>(ffn_w1, w1T, DMODEL, FDIM);
    transpose_w_kernel<<<dim3(128, 32), tb, 0, stream>>>(ffn_w3, w3T, DMODEL, FDIM);
    transpose_w_kernel<<<dim3(32, 128), tb, 0, stream>>>(ffn_w2, w2T, FDIM, DMODEL);

    // AdaLN vector + modulated LN1
    ada_kernel<<<(BATCH * 6 * DMODEL) / 256, 256, 0, stream>>>(t_emb, ada_w, ada_bi, ada);
    ln_mod_kernel<<<MTOT, 256, 0, stream>>>(x, ln1_g, ln1_b, ada, 0 * DMODEL, 1 * DMODEL, nx);

    // Four attention branches
    for (int p = 0; p < 4; ++p) {
        const bf16* wq = wT + (size_t)(p * 4 + 0) * MB1;
        const bf16* wk = wT + (size_t)(p * 4 + 1) * MB1;
        const bf16* wv = wT + (size_t)(p * 4 + 2) * MB1;
        const bf16* wo = wT + (size_t)(p * 4 + 3) * MB1;
        gemm_kernel<1><<<dim3(8, 32), 256, 0, stream>>>(nx, wq, Qb, nullptr, nullptr, nullptr, 0, MTOT, DMODEL, DMODEL);
        gemm_kernel<1><<<dim3(8, 32), 256, 0, stream>>>(nx, wk, Kb, nullptr, nullptr, nullptr, 0, MTOT, DMODEL, DMODEL);
        gemm_kernel<1><<<dim3(8, 32), 256, 0, stream>>>(nx, wv, Vb, nullptr, nullptr, nullptr, 0, MTOT, DMODEL, DMODEL);
        vtranspose_kernel<<<dim3(32, 2, BATCH * NHEAD), tb, 0, stream>>>(Vb, Vtb);
        flash_kernel<<<512, 256, 0, stream>>>(Qb, Kb, Vtb, attn, p);
        gemm_kernel<1><<<dim3(8, 32), 256, 0, stream>>>(attn, wo, obr[p], nullptr, nullptr, nullptr, 0, MTOT, DMODEL, DMODEL);
    }

    // Gate softmax + fusion, fused projection with AdaLN-gated residual
    gatefuse_kernel<<<MTOT, 256, 0, stream>>>(obr[0], obr[1], obr[2], obr[3], fus_wg, fus_bg, fused);
    gemm_kernel<3><<<dim3(8, 32), 256, 0, stream>>>(fused, fuswoT, x1, nullptr, x, ada, 2 * DMODEL, MTOT, DMODEL, DMODEL);

    // SwiGLU FFN with AdaLN, final gated residual straight into d_out
    ln_mod_kernel<<<MTOT, 256, 0, stream>>>(x1, ln2_g, ln2_b, ada, 3 * DMODEL, 4 * DMODEL, nx2);
    gemm_kernel<0><<<dim3(32, 32), 256, 0, stream>>>(nx2, w1T, h1, nullptr, nullptr, nullptr, 0, MTOT, FDIM, DMODEL);
    gemm_kernel<2><<<dim3(32, 32), 256, 0, stream>>>(nx2, w3T, hh, h1, nullptr, nullptr, 0, MTOT, FDIM, DMODEL);
    gemm_kernel<3><<<dim3(8, 32), 256, 0, stream>>>(hh, w2T, out, nullptr, x1, ada, 5 * DMODEL, MTOT, DMODEL, FDIM);
}
